// RoIHead_75711683493963
// MI455X (gfx1250) — matrix operand order repaired and checked by reading, not compiled, not run on hardware
//
#include <hip/hip_runtime.h>

// ---------------------------------------------------------------------------
// Types for CDNA5 WMMA
// ---------------------------------------------------------------------------
typedef _Float16 half8  __attribute__((ext_vector_type(8)));
typedef _Float16 v16h   __attribute__((ext_vector_type(16)));
typedef float    v8f    __attribute__((ext_vector_type(8)));

// Problem constants (match reference)
#define BATCH      4
#define NPROP      512
#define NCH        256
#define FH         50
#define FW         50
#define HWSZ       (FH*FW)           // 2500
#define RES        7
#define SR         2
#define GPTS       (RES*SR)          // 14
#define DDIM       (NCH*RES*RES)     // 12544
#define MROWS      (BATCH*NPROP)     // 2048
#define HID        1024
#define NCLS       81
#define KCLS       80
#define NCLS_PAD   128               // multiple of 64 -> branch-free GEMM
#define NBOX       (NCLS*4)          // 324
#define NBOX_PAD   384               // multiple of 64
#define DETS       100
#define SCORE_T    0.05f
#define NMS_T      0.5f
#define BBOX_CLIP  4.135166556742356f   // log(1000/16)
#define ROI_SCALE  (1.0f/32.0f)

// ---------------------------------------------------------------------------
// 0) Feature transpose: (B,C,H,W) f32 -> (B,H*W,C) f32  (coalesced channel runs)
// ---------------------------------------------------------------------------
__global__ __launch_bounds__(256) void transpose_features(
    const float* __restrict__ src, float* __restrict__ dst)
{
    size_t t = (size_t)blockIdx.x * 256 + threadIdx.x;
    size_t total = (size_t)BATCH * NCH * HWSZ;
    if (t >= total) return;
    int b  = (int)(t / (NCH * HWSZ));
    int r  = (int)(t % (NCH * HWSZ));
    int c  = r / HWSZ;
    int hw = r % HWSZ;
    dst[(size_t)b * (HWSZ * NCH) + (size_t)hw * NCH + c] = src[t];
}

// ---------------------------------------------------------------------------
// 1a) Generic transpose + pad fp32 weight (K x N) -> f16 (Npad x K)
// ---------------------------------------------------------------------------
__global__ __launch_bounds__(256) void transpose_pad_f16(
    const float* __restrict__ src, _Float16* __restrict__ dst,
    int K, int N, int Npad)
{
    size_t t = (size_t)blockIdx.x * 256 + threadIdx.x;
    size_t total = (size_t)K * Npad;
    if (t >= total) return;
    int k = (int)(t / Npad);
    int n = (int)(t % Npad);
    float v = (n < N) ? src[(size_t)k * N + n] : 0.0f;
    dst[(size_t)n * K + k] = (_Float16)v;
}

// ---------------------------------------------------------------------------
// 1b) W6 transpose with K permuted to match pooled layout (d = out*256 + c):
//     dst[n][out*256+c] = W6[c*49+out][n]
// ---------------------------------------------------------------------------
__global__ __launch_bounds__(256) void transpose_w6_perm(
    const float* __restrict__ src, _Float16* __restrict__ dst)
{
    size_t t = (size_t)blockIdx.x * 256 + threadIdx.x;
    size_t total = (size_t)HID * DDIM;
    if (t >= total) return;
    int n   = (int)(t / DDIM);
    int d   = (int)(t % DDIM);
    int out = d >> 8;        // 0..48
    int c   = d & 255;       // 0..255
    dst[t] = (_Float16)src[(size_t)(c * (RES * RES) + out) * HID + n];
}

// ---------------------------------------------------------------------------
// 2) RoIAlign on transposed features: thread = channel, fully coalesced.
//    Writes f16 row with feature index out*256 + c (matches W6 permutation).
// ---------------------------------------------------------------------------
__global__ __launch_bounds__(256) void roi_align_kernel(
    const float* __restrict__ featT,     // (B, 2500, 256)
    const float* __restrict__ props,     // (B, 512, 4)
    _Float16* __restrict__ X0)           // (2048, 12544) permuted layout
{
    const int roi  = blockIdx.x;               // 0..2047
    const int bimg = roi >> 9;
    const int tid  = threadIdx.x;              // == channel

    __shared__ int   s_y0[GPTS], s_y1[GPTS], s_x0[GPTS], s_x1[GPTS];
    __shared__ float s_ly[GPTS], s_lx[GPTS];
    __shared__ unsigned char s_vy[GPTS], s_vx[GPTS];

    const float* box = props + (size_t)roi * 4;
    float bx1 = box[0] * ROI_SCALE;
    float by1 = box[1] * ROI_SCALE;
    float bx2 = box[2] * ROI_SCALE;
    float by2 = box[3] * ROI_SCALE;
    float bw = fmaxf(bx2 - bx1, 1.0f) / RES;
    float bh = fmaxf(by2 - by1, 1.0f) / RES;

    if (tid < GPTS) {
        int g = tid;
        float off = ((float)g + 0.5f) / (float)SR;
        float yy = by1 + off * bh;
        s_vy[g] = (yy > -1.0f) && (yy < (float)FH);
        float y = fminf(fmaxf(yy, 0.0f), (float)(FH - 1));
        int y0 = (int)floorf(y);
        s_y0[g] = y0;
        s_y1[g] = min(y0 + 1, FH - 1);
        s_ly[g] = y - (float)y0;
    } else if (tid < 2 * GPTS) {
        int g = tid - GPTS;
        float off = ((float)g + 0.5f) / (float)SR;
        float xx = bx1 + off * bw;
        s_vx[g] = (xx > -1.0f) && (xx < (float)FW);
        float x = fminf(fmaxf(xx, 0.0f), (float)(FW - 1));
        int x0 = (int)floorf(x);
        s_x0[g] = x0;
        s_x1[g] = min(x0 + 1, FW - 1);
        s_lx[g] = x - (float)x0;
    }
    __syncthreads();

    const float* fb = featT + (size_t)bimg * (HWSZ * NCH);
    _Float16* orow = X0 + (size_t)roi * DDIM;

    #pragma unroll 1
    for (int out = 0; out < RES * RES; ++out) {
        int py = out / RES;
        int px = out - py * RES;
        float sum = 0.0f;
        #pragma unroll
        for (int sy = 0; sy < SR; ++sy) {
            int gy = py * SR + sy;
            float ly = s_ly[gy], hy = 1.0f - ly;
            int y0 = s_y0[gy], y1 = s_y1[gy];
            #pragma unroll
            for (int sx = 0; sx < SR; ++sx) {
                int gx = px * SR + sx;
                if (s_vy[gy] && s_vx[gx]) {
                    float lx = s_lx[gx], hx = 1.0f - lx;
                    int x0 = s_x0[gx], x1 = s_x1[gx];
                    sum += hy * hx * fb[(size_t)(y0 * FW + x0) * NCH + tid]
                         + hy * lx * fb[(size_t)(y0 * FW + x1) * NCH + tid]
                         + ly * hx * fb[(size_t)(y1 * FW + x0) * NCH + tid]
                         + ly * lx * fb[(size_t)(y1 * FW + x1) * NCH + tid];
                }
            }
        }
        orow[out * NCH + tid] = (_Float16)(sum * 0.25f);
    }
}

// ---------------------------------------------------------------------------
// 3) WMMA GEMM, branch-free inner loop, ping-pong double buffered (no register
//    rotation copies, no WAR v_nops).
//    Block: 8 waves x (32 rows) = 256 rows; 64 cols per block.
//    Wave tile 32x64 = 2x4 WMMA accumulators, K step 32, unrolled by 2.
//    Requires: M % 256 == 0, Npad % 64 == 0, K % 64 == 0, K >= 64.
// ---------------------------------------------------------------------------
__device__ __forceinline__ v16h load_a_frag(const _Float16* __restrict__ rowp,
                                            int k0, int hi)
{
    half8 lo = *(const half8*)(rowp + k0 + hi * 16);
    half8 hv = *(const half8*)(rowp + k0 + hi * 16 + 8);
    return __builtin_shufflevector(lo, hv, 0,1,2,3,4,5,6,7,8,9,10,11,12,13,14,15);
}

#define WMMA8(A0, A1, B0, B1, B2, B3)                                                                     \
    acc[0][0] = __builtin_amdgcn_wmma_f32_16x16x32_f16(false, A0, false, B0, (short)0, acc[0][0], false, false); \
    acc[1][0] = __builtin_amdgcn_wmma_f32_16x16x32_f16(false, A1, false, B0, (short)0, acc[1][0], false, false); \
    acc[0][1] = __builtin_amdgcn_wmma_f32_16x16x32_f16(false, A0, false, B1, (short)0, acc[0][1], false, false); \
    acc[1][1] = __builtin_amdgcn_wmma_f32_16x16x32_f16(false, A1, false, B1, (short)0, acc[1][1], false, false); \
    acc[0][2] = __builtin_amdgcn_wmma_f32_16x16x32_f16(false, A0, false, B2, (short)0, acc[0][2], false, false); \
    acc[1][2] = __builtin_amdgcn_wmma_f32_16x16x32_f16(false, A1, false, B2, (short)0, acc[1][2], false, false); \
    acc[0][3] = __builtin_amdgcn_wmma_f32_16x16x32_f16(false, A0, false, B3, (short)0, acc[0][3], false, false); \
    acc[1][3] = __builtin_amdgcn_wmma_f32_16x16x32_f16(false, A1, false, B3, (short)0, acc[1][3], false, false)

__global__ __launch_bounds__(256) void gemm_wmma_kernel(
    const _Float16* __restrict__ X,   // M x K (row-major)
    const _Float16* __restrict__ Wt,  // Npad x K (row-major)
    const float*    __restrict__ bias,
    int K, int Npad, int Nvalid,
    _Float16* __restrict__ Yh,        // optional f16 out (ld = Npad)
    float*    __restrict__ Yf,        // optional f32 out (ld = Npad)
    int relu)
{
    const int lane = threadIdx.x & 31;
    const int wave = threadIdx.x >> 5;
    const int r    = lane & 15;
    const int hi   = lane >> 4;
    const int m0   = blockIdx.y * 256 + wave * 32;
    const int n0   = blockIdx.x * 64;

    v8f acc[2][4];
    #pragma unroll
    for (int mt = 0; mt < 2; ++mt)
        #pragma unroll
        for (int nt = 0; nt < 4; ++nt)
            acc[mt][nt] = (v8f){0.f,0.f,0.f,0.f,0.f,0.f,0.f,0.f};

    const _Float16* a0p = X + (size_t)(m0 + r) * K;
    const _Float16* a1p = X + (size_t)(m0 + 16 + r) * K;
    const _Float16* b0p = Wt + (size_t)(n0 + r) * K;
    const _Float16* b1p = b0p + (size_t)16 * K;
    const _Float16* b2p = b0p + (size_t)32 * K;
    const _Float16* b3p = b0p + (size_t)48 * K;
    const int ho = hi * 16;

    // Set A: fragments for even K-steps; reloaded in place each iteration.
    v16h a0A = load_a_frag(a0p, 0, hi);
    v16h a1A = load_a_frag(a1p, 0, hi);
    v16h b0A = *(const v16h*)(b0p + ho);
    v16h b1A = *(const v16h*)(b1p + ho);
    v16h b2A = *(const v16h*)(b2p + ho);
    v16h b3A = *(const v16h*)(b3p + ho);

    int k0 = 0;
    #pragma unroll 1
    for (; k0 + 64 < K; k0 += 64) {
        // Set B: odd K-step fragments, produced and consumed this iteration.
        v16h a0B = load_a_frag(a0p, k0 + 32, hi);
        v16h a1B = load_a_frag(a1p, k0 + 32, hi);
        v16h b0B = *(const v16h*)(b0p + k0 + 32 + ho);
        v16h b1B = *(const v16h*)(b1p + k0 + 32 + ho);
        v16h b2B = *(const v16h*)(b2p + k0 + 32 + ho);
        v16h b3B = *(const v16h*)(b3p + k0 + 32 + ho);

        WMMA8(a0A, a1A, b0A, b1A, b2A, b3A);

        // Reload set A for next iteration (old/new live ranges disjoint).
        a0A = load_a_frag(a0p, k0 + 64, hi);
        a1A = load_a_frag(a1p, k0 + 64, hi);
        b0A = *(const v16h*)(b0p + k0 + 64 + ho);
        b1A = *(const v16h*)(b1p + k0 + 64 + ho);
        b2A = *(const v16h*)(b2p + k0 + 64 + ho);
        b3A = *(const v16h*)(b3p + k0 + 64 + ho);

        WMMA8(a0B, a1B, b0B, b1B, b2B, b3B);
    }
    // Tail: k0 == K-64; set A holds k0 fragments.
    {
        v16h a0B = load_a_frag(a0p, k0 + 32, hi);
        v16h a1B = load_a_frag(a1p, k0 + 32, hi);
        v16h b0B = *(const v16h*)(b0p + k0 + 32 + ho);
        v16h b1B = *(const v16h*)(b1p + k0 + 32 + ho);
        v16h b2B = *(const v16h*)(b2p + k0 + 32 + ho);
        v16h b3B = *(const v16h*)(b3p + k0 + 32 + ho);
        WMMA8(a0A, a1A, b0A, b1A, b2A, b3A);
        WMMA8(a0B, a1B, b0B, b1B, b2B, b3B);
    }

    // Store: element e -> row m0+mt*16+hi*8+e, col n0+nt*16+r
    #pragma unroll
    for (int mt = 0; mt < 2; ++mt) {
        #pragma unroll
        for (int nt = 0; nt < 4; ++nt) {
            int n = n0 + nt * 16 + r;
            float bv = (bias && n < Nvalid) ? bias[n] : 0.0f;
            #pragma unroll
            for (int e = 0; e < 8; ++e) {
                int m = m0 + mt * 16 + hi * 8 + e;
                float v = acc[mt][nt][e] + bv;
                if (relu) v = fmaxf(v, 0.0f);
                size_t o = (size_t)m * Npad + n;
                if (Yh) Yh[o] = (_Float16)v;
                if (Yf) Yf[o] = v;
            }
        }
    }
}

// ---------------------------------------------------------------------------
// 4) Softmax over 81 logits (ld=NCLS_PAD), scatter classes 1..80 to (B,80,512)
// ---------------------------------------------------------------------------
__global__ __launch_bounds__(256) void softmax_scores_kernel(
    const float* __restrict__ logits,   // 2048 x NCLS_PAD
    float* __restrict__ scores)         // (B*80) x 512
{
    int wave = threadIdx.x >> 5;
    int lane = threadIdx.x & 31;
    int row  = blockIdx.x * 8 + wave;
    if (row >= MROWS) return;
    const float* lg = logits + (size_t)row * NCLS_PAD;

    float mx = -3.4e38f;
    for (int i = lane; i < NCLS; i += 32) mx = fmaxf(mx, lg[i]);
    #pragma unroll
    for (int m = 16; m; m >>= 1) mx = fmaxf(mx, __shfl_xor(mx, m, 32));

    float sum = 0.0f;
    for (int i = lane; i < NCLS; i += 32) sum += __expf(lg[i] - mx);
    #pragma unroll
    for (int m = 16; m; m >>= 1) sum += __shfl_xor(sum, m, 32);

    float inv = 1.0f / sum;
    int b = row >> 9, n = row & 511;
    for (int c = lane; c < KCLS; c += 32) {
        scores[((size_t)(b * KCLS + c)) * NPROP + n] = __expf(lg[c + 1] - mx) * inv;
    }
}

// ---------------------------------------------------------------------------
// 5) Box decode + clip -> boxes (B,80,512,4)
// ---------------------------------------------------------------------------
__global__ __launch_bounds__(256) void decode_kernel(
    const float* __restrict__ props,    // (B,512,4)
    const float* __restrict__ deltas,   // 2048 x NBOX_PAD
    const int*   __restrict__ imgsz,
    float* __restrict__ boxes)          // (B*80*512) x 4
{
    int t = blockIdx.x * blockDim.x + threadIdx.x;
    if (t >= BATCH * KCLS * NPROP) return;
    int n = t & 511;
    int k = (t >> 9) % KCLS;
    int b = t / (KCLS * NPROP);

    const float* pb = props + ((size_t)(b * NPROP + n)) * 4;
    const float* dd = deltas + ((size_t)(b * NPROP + n)) * NBOX_PAD + (k + 1) * 4;
    float imax = (float)imgsz[0];

    float w  = pb[2] - pb[0];
    float h  = pb[3] - pb[1];
    float cx = pb[0] + 0.5f * w;
    float cy = pb[1] + 0.5f * h;
    float dx = dd[0], dy = dd[1];
    float dw = fminf(dd[2], BBOX_CLIP);
    float dh = fminf(dd[3], BBOX_CLIP);
    float pcx = dx * w + cx;
    float pcy = dy * h + cy;
    float pw  = __expf(dw) * w;
    float ph  = __expf(dh) * h;

    float x1 = fminf(fmaxf(pcx - 0.5f * pw, 0.0f), imax);
    float y1 = fminf(fmaxf(pcy - 0.5f * ph, 0.0f), imax);
    float x2 = fminf(fmaxf(pcx + 0.5f * pw, 0.0f), imax);
    float y2 = fminf(fmaxf(pcy + 0.5f * ph, 0.0f), imax);

    float* o = boxes + (size_t)t * 4;
    o[0] = x1; o[1] = y1; o[2] = x2; o[3] = y2;
}

// ---------------------------------------------------------------------------
// 6) NMS + top-100 per (b,class). One block of 256 threads per problem.
// ---------------------------------------------------------------------------
__global__ __launch_bounds__(256) void nms_kernel(
    const float* __restrict__ boxes,    // (320,512,4)
    const float* __restrict__ scores,   // (320,512)
    float* __restrict__ out)            // nb | ns | labels
{
    const int bk  = blockIdx.x;          // b*80 + k
    const int k   = bk % KCLS;
    const int tid = threadIdx.x;

    __shared__ float rsc[NPROP];
    __shared__ float rbx[NPROP * 4];
    __shared__ int   ord[NPROP];
    __shared__ float ssc[NPROP];
    __shared__ float sbx[NPROP * 4];
    __shared__ float area[NPROP];
    __shared__ unsigned char sv[NPROP];
    __shared__ unsigned char supp[NPROP];

    const float* bsrc = boxes + (size_t)bk * NPROP * 4;
    const float* ssrc = scores + (size_t)bk * NPROP;

    for (int i = tid; i < NPROP; i += 256) {
        rsc[i] = ssrc[i];
        float4 bx = *(const float4*)(bsrc + i * 4);
        rbx[i*4+0] = bx.x; rbx[i*4+1] = bx.y; rbx[i*4+2] = bx.z; rbx[i*4+3] = bx.w;
    }
    __syncthreads();

    // stable rank sort by descending score (matches argsort(-s))
    for (int i = tid; i < NPROP; i += 256) {
        float si = rsc[i];
        int rk = 0;
        for (int j = 0; j < NPROP; ++j) {
            float sj = rsc[j];
            rk += (sj > si) || ((sj == si) && (j < i));
        }
        ord[rk] = i;
    }
    __syncthreads();

    for (int i = tid; i < NPROP; i += 256) {
        int s = ord[i];
        float sc = rsc[s];
        ssc[i] = sc;
        float x1 = rbx[s*4+0], y1 = rbx[s*4+1], x2 = rbx[s*4+2], y2 = rbx[s*4+3];
        sbx[i*4+0] = x1; sbx[i*4+1] = y1; sbx[i*4+2] = x2; sbx[i*4+3] = y2;
        area[i] = fmaxf(x2 - x1, 0.0f) * fmaxf(y2 - y1, 0.0f);
        sv[i]   = (sc > SCORE_T) ? 1 : 0;
        supp[i] = 0;
    }
    __syncthreads();

    for (int i = 0; i < NPROP; ++i) {
        bool keep_i = sv[i] && !supp[i];
        if (keep_i) {
            float xi1 = sbx[i*4+0], yi1 = sbx[i*4+1];
            float xi2 = sbx[i*4+2], yi2 = sbx[i*4+3];
            float ai  = area[i];
            for (int j = i + 1 + tid; j < NPROP; j += 256) {
                float lx = fmaxf(xi1, sbx[j*4+0]);
                float ly = fmaxf(yi1, sbx[j*4+1]);
                float rx = fminf(xi2, sbx[j*4+2]);
                float ry = fminf(yi2, sbx[j*4+3]);
                float iw = fmaxf(rx - lx, 0.0f);
                float ih = fmaxf(ry - ly, 0.0f);
                float inter = iw * ih;
                float iou = inter / (ai + area[j] - inter + 1e-9f);
                if (iou > NMS_T) supp[j] = 1;
            }
        }
        __syncthreads();
    }

    float* nb = out + (size_t)bk * DETS * 4;
    float* ns = out + (size_t)(BATCH * KCLS * DETS * 4) + (size_t)bk * DETS;
    float* lb = out + (size_t)(BATCH * KCLS * DETS * 5) + (size_t)bk * DETS;
    if (tid < DETS) lb[tid] = (float)(k + 1);

    if (tid == 0) {
        int cnt = 0;
        for (int i = 0; i < NPROP && cnt < DETS; ++i) {
            if (sv[i] && !supp[i]) {
                nb[cnt*4+0] = sbx[i*4+0]; nb[cnt*4+1] = sbx[i*4+1];
                nb[cnt*4+2] = sbx[i*4+2]; nb[cnt*4+3] = sbx[i*4+3];
                ns[cnt] = ssc[i];
                ++cnt;
            }
        }
        for (int i = 0; i < NPROP && cnt < DETS; ++i) {
            if (!(sv[i] && !supp[i])) {
                nb[cnt*4+0] = sbx[i*4+0]; nb[cnt*4+1] = sbx[i*4+1];
                nb[cnt*4+2] = sbx[i*4+2]; nb[cnt*4+3] = sbx[i*4+3];
                ns[cnt] = -1.0f;
                ++cnt;
            }
        }
    }
}

// ---------------------------------------------------------------------------
// Host launcher
// ---------------------------------------------------------------------------
extern "C" void kernel_launch(void* const* d_in, const int* in_sizes, int n_in,
                              void* d_out, int out_size, void* d_ws, size_t ws_size,
                              hipStream_t stream)
{
    const float* features  = (const float*)d_in[0];
    const float* proposals = (const float*)d_in[1];
    const float* W6 = (const float*)d_in[2];
    const float* b6 = (const float*)d_in[3];
    const float* W7 = (const float*)d_in[4];
    const float* b7 = (const float*)d_in[5];
    const float* Wc = (const float*)d_in[6];
    const float* bc = (const float*)d_in[7];
    const float* Wb = (const float*)d_in[8];
    const float* bb = (const float*)d_in[9];
    const int* imgsz = (const int*)d_in[10];
    float* out = (float*)d_out;

    char* ws = (char*)d_ws;
    size_t off = 0;
    auto take = [&](size_t bytes) -> char* {
        char* p = ws + off;
        off += (bytes + 255) & ~(size_t)255;
        return p;
    };

    float*    featT = (float*)take((size_t)BATCH * HWSZ * NCH * 4);
    _Float16* W6t = (_Float16*)take((size_t)HID * DDIM * 2);
    _Float16* W7t = (_Float16*)take((size_t)HID * HID * 2);
    _Float16* Wct = (_Float16*)take((size_t)NCLS_PAD * HID * 2);
    _Float16* Wbt = (_Float16*)take((size_t)NBOX_PAD * HID * 2);
    _Float16* X0  = (_Float16*)take((size_t)MROWS * DDIM * 2);
    _Float16* X1  = (_Float16*)take((size_t)MROWS * HID * 2);
    _Float16* X2  = (_Float16*)take((size_t)MROWS * HID * 2);
    float* logits = (float*)take((size_t)MROWS * NCLS_PAD * 4);
    float* deltas = (float*)take((size_t)MROWS * NBOX_PAD * 4);
    float* scores = (float*)take((size_t)BATCH * KCLS * NPROP * 4);
    float* dboxes = (float*)take((size_t)BATCH * KCLS * NPROP * 4 * 4);
    (void)ws_size; (void)in_sizes; (void)n_in; (void)out_size;

    // Feature transpose (B,C,H,W) -> (B,HW,C)
    {
        size_t tot = (size_t)BATCH * NCH * HWSZ;
        transpose_features<<<(unsigned)((tot + 255) / 256), 256, 0, stream>>>(
            features, featT);
    }

    // Weight conversion / transpose
    {
        size_t t6 = (size_t)HID * DDIM;
        transpose_w6_perm<<<(unsigned)((t6 + 255) / 256), 256, 0, stream>>>(W6, W6t);
        size_t t7 = (size_t)HID * HID;
        transpose_pad_f16<<<(unsigned)((t7 + 255) / 256), 256, 0, stream>>>(
            W7, W7t, HID, HID, HID);
        size_t tc = (size_t)HID * NCLS_PAD;
        transpose_pad_f16<<<(unsigned)((tc + 255) / 256), 256, 0, stream>>>(
            Wc, Wct, HID, NCLS, NCLS_PAD);
        size_t tb = (size_t)HID * NBOX_PAD;
        transpose_pad_f16<<<(unsigned)((tb + 255) / 256), 256, 0, stream>>>(
            Wb, Wbt, HID, NBOX, NBOX_PAD);
    }

    // RoIAlign -> X0 (f16, permuted feature order)
    roi_align_kernel<<<MROWS, 256, 0, stream>>>(featT, proposals, X0);

    // FC6: X0 (2048x12544) * W6t^T -> relu -> X1 (f16)
    {
        dim3 g(HID / 64, MROWS / 256);
        gemm_wmma_kernel<<<g, 256, 0, stream>>>(
            X0, W6t, b6, DDIM, HID, HID, X1, nullptr, 1);
    }
    // FC7: X1 * W7t^T -> relu -> X2 (f16)
    {
        dim3 g(HID / 64, MROWS / 256);
        gemm_wmma_kernel<<<g, 256, 0, stream>>>(
            X1, W7t, b7, HID, HID, HID, X2, nullptr, 1);
    }
    // Class head: X2 * Wct^T -> logits (f32, ld 128)
    {
        dim3 g(NCLS_PAD / 64, MROWS / 256);
        gemm_wmma_kernel<<<g, 256, 0, stream>>>(
            X2, Wct, bc, HID, NCLS_PAD, NCLS, nullptr, logits, 0);
    }
    // Box head: X2 * Wbt^T -> deltas (f32, ld 384)
    {
        dim3 g(NBOX_PAD / 64, MROWS / 256);
        gemm_wmma_kernel<<<g, 256, 0, stream>>>(
            X2, Wbt, bb, HID, NBOX_PAD, NBOX, nullptr, deltas, 0);
    }

    // Softmax -> scores (B,80,512)
    softmax_scores_kernel<<<MROWS / 8, 256, 0, stream>>>(logits, scores);

    // Decode -> boxes (B,80,512,4)
    {
        int tot = BATCH * KCLS * NPROP;
        decode_kernel<<<(tot + 255) / 256, 256, 0, stream>>>(
            proposals, deltas, imgsz, dboxes);
    }

    // NMS + top-100 per (b,k)
    nms_kernel<<<BATCH * KCLS, 256, 0, stream>>>(dboxes, scores, out);
}
